// Decoder_8194797601152
// MI455X (gfx1250) — compile-verified, matrix operand order checked by reading, not run on hardware
//
#include <hip/hip_runtime.h>
#include <hip/hip_bf16.h>
#include <math.h>

#define BATCH    64
#define EMBED    512
#define HIDDEN   1024
#define VOCABN   32000
#define MAXVOCAB 33000
#define MAXLEN   128

typedef __attribute__((ext_vector_type(16))) __bf16 v16bf;
typedef __attribute__((ext_vector_type(8)))  float  v8f;

// Pack one WMMA 16x16x32 bf16 operand fragment from an f32 K-contiguous row.
// Per CDNA5 ISA (CDNA5 ISA, 16-bit A 16x32): lane half = lane>>4;
// elements 0..7  <- K = k0 + half*8 + (0..7)
// elements 8..15 <- K = k0 + 16 + half*8 + (0..7)
__device__ __forceinline__ v16bf load_frag_bf16(const float* __restrict__ rowk, int half) {
  const float4 q0 = *(const float4*)(rowk + half * 8);
  const float4 q1 = *(const float4*)(rowk + half * 8 + 4);
  const float4 q2 = *(const float4*)(rowk + half * 8 + 16);
  const float4 q3 = *(const float4*)(rowk + half * 8 + 20);
  v16bf f;
  f[0]  = (__bf16)q0.x; f[1]  = (__bf16)q0.y; f[2]  = (__bf16)q0.z; f[3]  = (__bf16)q0.w;
  f[4]  = (__bf16)q1.x; f[5]  = (__bf16)q1.y; f[6]  = (__bf16)q1.z; f[7]  = (__bf16)q1.w;
  f[8]  = (__bf16)q2.x; f[9]  = (__bf16)q2.y; f[10] = (__bf16)q2.z; f[11] = (__bf16)q2.w;
  f[12] = (__bf16)q3.x; f[13] = (__bf16)q3.y; f[14] = (__bf16)q3.z; f[15] = (__bf16)q3.w;
  return f;
}

#define WMMA_BF16(A, B, C) \
  __builtin_amdgcn_wmma_f32_16x16x32_bf16(false, (A), false, (B), (short)0, (C), false, false)

// ---------------------------------------------------------------------------
// C[64,N] = act(A[64,K] @ B[N,K]^T + bias).  One wave per 16-wide N tile,
// 4 accumulators covering all of M=64 so B (e.g. Wo_W, 131MB) streams once.
// ---------------------------------------------------------------------------
__global__ void gemm64_bt_wmma(const float* __restrict__ A, int lda,
                               const float* __restrict__ B, int ldb,
                               const float* __restrict__ bias,
                               float* __restrict__ C, int ldc,
                               int N, int K, int relu) {
  const int lane  = threadIdx.x & 31;
  const int wid   = threadIdx.x >> 5;
  const int tileN = blockIdx.x * (blockDim.x >> 5) + wid;
  if (tileN * 16 >= N) return;                    // wave-uniform: EXEC stays all-ones
  const int half = lane >> 4;
  const int l15  = lane & 15;
  const int n    = tileN * 16 + l15;

  v8f acc0 = {}, acc1 = {}, acc2 = {}, acc3 = {};
  const float* __restrict__ brow = B + (size_t)n * ldb;
  for (int k0 = 0; k0 < K; k0 += 32) {
    v16bf bf = load_frag_bf16(brow + k0, half);
    v16bf a0 = load_frag_bf16(A + (size_t)(l15)      * lda + k0, half);
    v16bf a1 = load_frag_bf16(A + (size_t)(16 + l15) * lda + k0, half);
    v16bf a2 = load_frag_bf16(A + (size_t)(32 + l15) * lda + k0, half);
    v16bf a3 = load_frag_bf16(A + (size_t)(48 + l15) * lda + k0, half);
    acc0 = WMMA_BF16(a0, bf, acc0);
    acc1 = WMMA_BF16(a1, bf, acc1);
    acc2 = WMMA_BF16(a2, bf, acc2);
    acc3 = WMMA_BF16(a3, bf, acc3);
  }
  const float bv = bias ? bias[n] : 0.0f;
#pragma unroll
  for (int r = 0; r < 8; ++r) {
    const int rr = half * 8 + r;                 // C layout: VGPR r -> M=r (+8 for upper half)
    float v;
    v = acc0[r] + bv; if (relu) v = fmaxf(v, 0.0f); C[(size_t)(rr)      * ldc + n] = v;
    v = acc1[r] + bv; if (relu) v = fmaxf(v, 0.0f); C[(size_t)(16 + rr) * ldc + n] = v;
    v = acc2[r] + bv; if (relu) v = fmaxf(v, 0.0f); C[(size_t)(32 + rr) * ldc + n] = v;
    v = acc3[r] + bv; if (relu) v = fmaxf(v, 0.0f); C[(size_t)(48 + rr) * ldc + n] = v;
  }
}

// ---------------------------------------------------------------------------
// Fused: score_c[b,l] = sum_h tanh( enc[b,l,:] @ Wc^T + Wc_b )[h] * h_new[b,h]
// One wave per 16 enc rows (rows never straddle a batch: 128 rows/batch = 8 tiles),
// loops over all 64 h-tiles, tanh+weight+lane-shuffle-reduce, no atomics.
// ---------------------------------------------------------------------------
__global__ void scorec_wmma(const float* __restrict__ enc,   // [64*128, 1024]
                            const float* __restrict__ Wc,    // [1024, 1024]
                            const float* __restrict__ Wcb,   // [1024]
                            const float* __restrict__ hnew,  // [64, 1024]
                            float* __restrict__ sc) {        // [64*128]
  const int lane  = threadIdx.x & 31;
  const int wid   = threadIdx.x >> 5;
  const int mtile = blockIdx.x * (blockDim.x >> 5) + wid;    // 0..511
  if (mtile >= (BATCH * MAXLEN) / 16) return;
  const int half = lane >> 4;
  const int l15  = lane & 15;
  const int b    = mtile >> 3;                               // 8 mtiles per batch row
  const float* __restrict__ arow = enc + (size_t)(mtile * 16 + l15) * HIDDEN;

  float rsum[8];
#pragma unroll
  for (int r = 0; r < 8; ++r) rsum[r] = 0.0f;

  for (int nt = 0; nt < HIDDEN / 16; ++nt) {
    const int n = nt * 16 + l15;
    const float* __restrict__ brow = Wc + (size_t)n * HIDDEN;
    v8f acc = {};
    for (int k0 = 0; k0 < HIDDEN; k0 += 32) {
      v16bf af = load_frag_bf16(arow + k0, half);
      v16bf bf = load_frag_bf16(brow + k0, half);
      acc = WMMA_BF16(af, bf, acc);
    }
    const float cb = Wcb[n];
    const float hb = hnew[(size_t)b * HIDDEN + n];
#pragma unroll
    for (int r = 0; r < 8; ++r) rsum[r] += tanhf(acc[r] + cb) * hb;
  }
  // Reduce over the 16 lanes of each half (cols) -> per-row scalar.
#pragma unroll
  for (int r = 0; r < 8; ++r) {
    float v = rsum[r];
    v += __shfl_xor(v, 1, 32);
    v += __shfl_xor(v, 2, 32);
    v += __shfl_xor(v, 4, 32);
    v += __shfl_xor(v, 8, 32);
    if (l15 == 0) sc[mtile * 16 + half * 8 + r] = v;         // row index == b*128 + l
  }
}

// ---------------------------------------------------------------------------
// Per-batch setup: embedding gather, sel_reading (trunc), attention logits +
// softmax + context.  Writes cat1=[emb|attn_applied], gru_in[:,1024:]=sel_reading,
// and the attn_weights output.
// ---------------------------------------------------------------------------
__global__ void setup_kernel(const int* __restrict__ tok, const float* __restrict__ hidden,
                             const float* __restrict__ enc, const int* __restrict__ seq,
                             const float* __restrict__ pre, const float* __restrict__ emb,
                             const float* __restrict__ attnW, const float* __restrict__ attnb,
                             float* __restrict__ cat1, float* __restrict__ gru_in,
                             float* __restrict__ attn_out) {
  __shared__ float s_emb[EMBED];
  __shared__ float s_h[HIDDEN];
  __shared__ float s_sel[MAXLEN];
  __shared__ float s_w[MAXLEN];
  __shared__ float s_red[2];
  const int b = blockIdx.x;
  const int t = threadIdx.x;                                  // 256 threads
  const int tk = tok[b];

  for (int i = t; i < EMBED; i += 256) {
    float e = emb[(size_t)tk * EMBED + i];
    s_emb[i] = e;
    cat1[(size_t)b * (EMBED + HIDDEN) + i] = e;
  }
  for (int i = t; i < HIDDEN; i += 256) s_h[i] = hidden[(size_t)b * HIDDEN + i];
  if (t < MAXLEN) s_sel[t] = (seq[b * MAXLEN + t] == tk) ? pre[b * MAXLEN + t] : 0.0f;
  __syncthreads();

  const float* __restrict__ encb = enc + (size_t)b * MAXLEN * HIDDEN;
  for (int h = t; h < HIDDEN; h += 256) {                     // sel_reading
    float s = 0.0f;
    for (int l = 0; l < MAXLEN; ++l) s += s_sel[l] * truncf(encb[(size_t)l * HIDDEN + h]);
    gru_in[(size_t)b * (2 * HIDDEN) + HIDDEN + h] = s;
  }
  if (t < MAXLEN) {                                           // attention logits
    const float* __restrict__ w = attnW + (size_t)t * (EMBED + HIDDEN);
    float a = attnb[t];
    for (int k = 0; k < EMBED; ++k)  a += s_emb[k] * w[k];
    for (int k = 0; k < HIDDEN; ++k) a += s_h[k] * w[EMBED + k];
    s_w[t] = a;
  }
  __syncthreads();
  if (t == 0) {
    float m = -1e30f;
    for (int l = 0; l < MAXLEN; ++l) m = fmaxf(m, s_w[l]);
    float s = 0.0f;
    for (int l = 0; l < MAXLEN; ++l) s += expf(s_w[l] - m);
    s_red[0] = m; s_red[1] = 1.0f / s;
  }
  __syncthreads();
  if (t < MAXLEN) {
    float w = expf(s_w[t] - s_red[0]) * s_red[1];
    s_w[t] = w;
    attn_out[b * MAXLEN + t] = w;
  }
  __syncthreads();
  for (int h = t; h < HIDDEN; h += 256) {                     // attn_applied
    float s = 0.0f;
    for (int l = 0; l < MAXLEN; ++l) s += s_w[l] * encb[(size_t)l * HIDDEN + h];
    cat1[(size_t)b * (EMBED + HIDDEN) + EMBED + h] = s;
  }
}

// GRU gates -> h_new
__global__ void gru_kernel(const float* __restrict__ gi, const float* __restrict__ gh,
                           const float* __restrict__ hidden, float* __restrict__ hnew) {
  const int idx = blockIdx.x * blockDim.x + threadIdx.x;
  if (idx >= BATCH * HIDDEN) return;
  const int b = idx / HIDDEN, h = idx % HIDDEN;
  const float* gib = gi + (size_t)b * 3 * HIDDEN;
  const float* ghb = gh + (size_t)b * 3 * HIDDEN;
  const float r = 1.0f / (1.0f + expf(-(gib[h] + ghb[h])));
  const float z = 1.0f / (1.0f + expf(-(gib[HIDDEN + h] + ghb[HIDDEN + h])));
  const float n = tanhf(gib[2 * HIDDEN + h] + r * ghb[2 * HIDDEN + h]);
  hnew[idx] = (1.0f - z) * n + z * hidden[idx];
}

// Softmax over 32128, scatter-add copy-probs, where/log -> output1 + prob_c
__global__ void final_kernel(const float* __restrict__ score_g, const float* __restrict__ score_c,
                             const int* __restrict__ seq, float* __restrict__ out1,
                             float* __restrict__ probc_out) {
  __shared__ float sred[256];
  __shared__ float s_pc[MAXLEN];
  const int b = blockIdx.x, t = threadIdx.x;
  const float* __restrict__ sg = score_g + (size_t)b * VOCABN;
  const float* __restrict__ sc = score_c + b * MAXLEN;

  float m = -1e30f;
  for (int v = t; v < VOCABN; v += 256) m = fmaxf(m, sg[v]);
  if (t < MAXLEN) m = fmaxf(m, sc[t]);
  sred[t] = m; __syncthreads();
  for (int s = 128; s > 0; s >>= 1) { if (t < s) sred[t] = fmaxf(sred[t], sred[t + s]); __syncthreads(); }
  m = sred[0]; __syncthreads();

  float sum = 0.0f;
  for (int v = t; v < VOCABN; v += 256) sum += expf(sg[v] - m);
  if (t < MAXLEN) sum += expf(sc[t] - m);
  sred[t] = sum; __syncthreads();
  for (int s = 128; s > 0; s >>= 1) { if (t < s) sred[t] += sred[t + s]; __syncthreads(); }
  const float inv = 1.0f / sred[0];

  float* __restrict__ row = out1 + (size_t)b * MAXVOCAB;
  for (int v = t; v < VOCABN; v += 256) row[v] = expf(sg[v] - m) * inv;
  for (int v = VOCABN + t; v < MAXVOCAB; v += 256) row[v] = 0.0f;
  if (t < MAXLEN) {
    float p = expf(sc[t] - m) * inv;
    s_pc[t] = p;
    probc_out[b * MAXLEN + t] = p;
  }
  __threadfence();
  __syncthreads();
  if (t == 0) {                                   // serial scatter: deterministic
    for (int l = 0; l < MAXLEN; ++l) row[seq[b * MAXLEN + l]] += s_pc[l];
  }
  __threadfence();
  __syncthreads();
  for (int v = t; v < MAXVOCAB; v += 256) {
    float x = row[v];
    if (x == 0.0f) x = 1e-9f;
    if (v == 2) x = 1e-9f;
    row[v] = logf(x);
  }
}

extern "C" void kernel_launch(void* const* d_in, const int* in_sizes, int n_in,
                              void* d_out, int out_size, void* d_ws, size_t ws_size,
                              hipStream_t stream) {
  const int*   input_tok = (const int*)d_in[0];
  const float* hidden    = (const float*)d_in[1];
  const float* enc       = (const float*)d_in[2];
  const int*   input_seq = (const int*)d_in[3];
  const float* pre_prob  = (const float*)d_in[4];
  const float* emb       = (const float*)d_in[5];
  const float* attn_W    = (const float*)d_in[6];
  const float* attn_b    = (const float*)d_in[7];
  const float* comb_W    = (const float*)d_in[8];
  const float* comb_b    = (const float*)d_in[9];
  const float* W_ih      = (const float*)d_in[10];
  const float* W_hh      = (const float*)d_in[11];
  const float* b_ih      = (const float*)d_in[12];
  const float* b_hh      = (const float*)d_in[13];
  const float* Wo_W      = (const float*)d_in[14];
  const float* Wo_b      = (const float*)d_in[15];
  const float* Wc_W      = (const float*)d_in[16];
  const float* Wc_b      = (const float*)d_in[17];

  // Output tuple, flat: output1[64,33000], h_new[1,64,1024], attn_w[64,128], prob_c[64,128]
  float* out1      = (float*)d_out;
  float* hnew      = out1 + (size_t)BATCH * MAXVOCAB;
  float* attn_out  = hnew + (size_t)BATCH * HIDDEN;
  float* probc_out = attn_out + (size_t)BATCH * MAXLEN;

  float* ws      = (float*)d_ws;
  float* cat1    = ws;  ws += (size_t)BATCH * (EMBED + HIDDEN);   // 64 x 1536
  float* gru_in  = ws;  ws += (size_t)BATCH * 2 * HIDDEN;         // 64 x 2048
  float* gi      = ws;  ws += (size_t)BATCH * 3 * HIDDEN;         // 64 x 3072
  float* gh      = ws;  ws += (size_t)BATCH * 3 * HIDDEN;         // 64 x 3072
  float* score_g = ws;  ws += (size_t)BATCH * VOCABN;             // 64 x 32000
  float* score_c = ws;  ws += (size_t)BATCH * MAXLEN;             // 64 x 128

  setup_kernel<<<BATCH, 256, 0, stream>>>(input_tok, hidden, enc, input_seq, pre_prob,
                                          emb, attn_W, attn_b, cat1, gru_in, attn_out);
  // out = relu(cat1 @ comb_W^T + b) -> gru_in[:, 0:1024] (ldc=2048)
  gemm64_bt_wmma<<<16, 128, 0, stream>>>(cat1, EMBED + HIDDEN, comb_W, EMBED + HIDDEN,
                                         comb_b, gru_in, 2 * HIDDEN, HIDDEN, EMBED + HIDDEN, 1);
  // gi = gru_in @ W_ih^T + b_ih   (64x3072, K=2048)
  gemm64_bt_wmma<<<48, 128, 0, stream>>>(gru_in, 2 * HIDDEN, W_ih, 2 * HIDDEN,
                                         b_ih, gi, 3 * HIDDEN, 3 * HIDDEN, 2 * HIDDEN, 0);
  // gh = h @ W_hh^T + b_hh        (64x3072, K=1024)
  gemm64_bt_wmma<<<48, 128, 0, stream>>>(hidden, HIDDEN, W_hh, HIDDEN,
                                         b_hh, gh, 3 * HIDDEN, 3 * HIDDEN, HIDDEN, 0);
  gru_kernel<<<(BATCH * HIDDEN + 255) / 256, 256, 0, stream>>>(gi, gh, hidden, hnew);
  // score_g = h_new @ Wo_W^T + Wo_b  (64x32000, K=1024) -- streams Wo once
  gemm64_bt_wmma<<<500, 128, 0, stream>>>(hnew, HIDDEN, Wo_W, HIDDEN,
                                          Wo_b, score_g, VOCABN, VOCABN, HIDDEN, 0);
  // fused tanh(enc @ Wc^T + b) . h_new  (17.2 GF on WMMA)
  scorec_wmma<<<128, 128, 0, stream>>>(enc, Wc_W, Wc_b, hnew, score_c);
  final_kernel<<<BATCH, 256, 0, stream>>>(score_g, score_c, input_seq, out1, probc_out);
}
